// AttentionPropagation_38448547234594
// MI455X (gfx1250) — matrix operand order repaired and checked by reading, not compiled, not run on hardware
//
#include <hip/hip_runtime.h>
#include <hip/hip_bf16.h>
#include <math.h>

typedef _Float16 half_t;
typedef __attribute__((ext_vector_type(16))) _Float16 v16h;
typedef __attribute__((ext_vector_type(8)))  float    v8f;
typedef __attribute__((ext_vector_type(4)))  float    v4f;

#define FEAT 128
#define B_   2
#define N1_  4096
#define N2_  32768
#define QT   64          // queries per block (4 waves x 16 rows)
#define KT   32          // keys per flash iteration
#define SCALE_ 0.08838834764831845f   // 128^-0.5

union F16x16 { v4f f4[2]; v16h h16; };

// ---------------------------------------------------------------------------
// Projection: out[row][f] = (bias[f] + xyz(3)·W[f][0:3] + feat(128)·W[f][3:131]) * scale
// ---------------------------------------------------------------------------
__global__ __launch_bounds__(128)
void proj131_kernel(const float* __restrict__ xyz, const float* __restrict__ feat,
                    const float* __restrict__ W, const float* __restrict__ bias,
                    half_t* __restrict__ out, float scale)
{
    const int row = blockIdx.x;
    const int f   = threadIdx.x;
    const float* x3 = xyz  + (size_t)row * 3;
    const float* xf = feat + (size_t)row * FEAT;
    const float* w  = W    + (size_t)f * (FEAT + 3);
    float acc = bias[f] + x3[0]*w[0] + x3[1]*w[1] + x3[2]*w[2];
    #pragma unroll 8
    for (int d = 0; d < FEAT; ++d) acc += xf[d] * w[3 + d];
    out[(size_t)row * FEAT + f] = (half_t)(acc * scale);
}

// V projection, stored transposed: outT[b][f][n1] so PV B-fragments are contiguous
__global__ __launch_bounds__(128)
void projv_kernel(const float* __restrict__ feat, const float* __restrict__ W,
                  const float* __restrict__ bias, half_t* __restrict__ outT)
{
    const int row = blockIdx.x;           // b*N1 + n
    const int f   = threadIdx.x;
    const int b = row / N1_, n = row % N1_;
    const float* xf = feat + (size_t)row * FEAT;
    const float* w  = W    + (size_t)f * FEAT;
    float acc = bias[f];
    #pragma unroll 8
    for (int d = 0; d < FEAT; ++d) acc += xf[d] * w[d];
    outT[((size_t)b * FEAT + f) * N1_ + n] = (half_t)acc;
}

// fp32 -> fp16 weight conversion (for fused output projection)
__global__ __launch_bounds__(256)
void cvt16_kernel(const float* __restrict__ src, half_t* __restrict__ dst, int n)
{
    const int i = blockIdx.x * 256 + threadIdx.x;
    if (i < n) dst[i] = (half_t)src[i];
}

// ---------------------------------------------------------------------------
// Flash attention with euclidean-distance penalty + fused output projection.
// Block = 128 threads = 4 wave32; each wave owns 16 query rows.
// ---------------------------------------------------------------------------
__global__ __launch_bounds__(128)
void attn_kernel(const half_t* __restrict__ Qh,    // [B*N2,128] (pre-scaled)
                 const half_t* __restrict__ Kh,    // [B*N1,128]
                 const half_t* __restrict__ VhT,   // [B,128,N1]
                 const half_t* __restrict__ Woh,   // [128,128] f16 (row f_out, col d)
                 const float*  __restrict__ bo,    // [128]
                 const float*  __restrict__ dxyz,  // [B,N2,3]
                 const float*  __restrict__ sxyz,  // [B,N1,3]
                 const float*  __restrict__ dfeat, // [B,N2,128]
                 float* __restrict__ out)          // [B,N2,128]
{
    const int nqb  = N2_ / QT;
    const int b    = blockIdx.x / nqb;
    const int q0   = (blockIdx.x % nqb) * QT;
    const int tid  = threadIdx.x;
    const int wave = tid >> 5;
    const int lane = tid & 31;
    const int lh   = lane & 15;
    const int hi   = (lane >> 4) & 1;

    __shared__ float  s_qxyz[QT * 3];
    __shared__ half_t s_p[4][16][KT + 8];      // per-wave P transpose strip
    __shared__ half_t s_x[4][16][FEAT + 8];    // per-wave epilogue X strip

    for (int i = tid; i < QT * 3; i += 128)
        s_qxyz[i] = dxyz[((size_t)b * N2_ + q0) * 3 + i];
    __syncthreads();

    // --- Q A-fragments (held for whole kernel). A 16x32 f16 layout:
    // lane(0-15): M=lane, K = 0..7 (v0-3) & 16..23 (v4-7); lane(16-31): K = 8..15 & 24..31
    const size_t qrow = (size_t)b * N2_ + q0 + wave * 16 + lh;
    const half_t* qptr = Qh + qrow * FEAT;
    v16h qa[4];
    #pragma unroll
    for (int c = 0; c < 4; ++c) {
        const int o0 = 32 * c + hi * 8;
        F16x16 u;
        u.f4[0] = *(const v4f*)(qptr + o0);
        u.f4[1] = *(const v4f*)(qptr + o0 + 16);
        qa[c] = u.h16;
    }

    // Per-C-slot query xyz: slot r covers row wave*16 + r + 8*hi
    float qx[8], qy[8], qz[8];
    #pragma unroll
    for (int r = 0; r < 8; ++r) {
        const int m = wave * 16 + r + hi * 8;
        qx[r] = s_qxyz[m * 3 + 0];
        qy[r] = s_qxyz[m * 3 + 1];
        qz[r] = s_qxyz[m * 3 + 2];
    }

    float mrow[8], lrow[8];
    v8f o[8];
    v8f vzero = {};
    #pragma unroll
    for (int r = 0; r < 8; ++r) { mrow[r] = -1e30f; lrow[r] = 0.0f; }
    #pragma unroll
    for (int ft = 0; ft < 8; ++ft) o[ft] = vzero;

    for (int kt = 0; kt < N1_ / KT; ++kt) {
        const int k0 = kt * KT;

        // ---- prefetch next key tile (K rows + V columns) into near caches
        if (kt + 1 < N1_ / KT) {
            const half_t* nk = Kh + ((size_t)b * N1_ + k0 + KT + lane) * FEAT;
            __builtin_prefetch(nk, 0, 3);
            __builtin_prefetch(nk + 64, 0, 3);
            #pragma unroll
            for (int j = 0; j < 4; ++j)
                __builtin_prefetch(VhT + ((size_t)b * FEAT + j * 32 + lane) * N1_ + k0 + KT, 0, 3);
        }

        v8f s[2];
        // ---- S = Q·K^T (scale folded into Q), minus 0.5*dist
        #pragma unroll
        for (int nt = 0; nt < 2; ++nt) {
            const size_t krow = (size_t)b * N1_ + k0 + nt * 16 + lh;
            const half_t* kptr = Kh + krow * FEAT;
            v8f acc = vzero;
            #pragma unroll
            for (int c = 0; c < 4; ++c) {
                // B 32x16 f16 layout: lanes 0-15 hold K=0..7 & 16..23, lanes 16-31 K=8..15 & 24..31; N=lane&15
                const int o0 = 32 * c + hi * 8;
                F16x16 ub;
                ub.f4[0] = *(const v4f*)(kptr + o0);
                ub.f4[1] = *(const v4f*)(kptr + o0 + 16);
                acc = __builtin_amdgcn_wmma_f32_16x16x32_f16(false, qa[c], false, ub.h16,
                                                             (short)0, acc, false, false);
            }
            const float* kp = sxyz + ((size_t)b * N1_ + k0 + nt * 16 + lh) * 3;
            const float kxx = kp[0], kxy = kp[1], kxz = kp[2];
            #pragma unroll
            for (int r = 0; r < 8; ++r) {
                const float dx = qx[r] - kxx, dy = qy[r] - kxy, dz = qz[r] - kxz;
                const float sq = dx*dx + dy*dy + dz*dz;
                acc[r] -= 0.5f * sqrtf(fmaxf(sq, 1e-12f));
            }
            s[nt] = acc;
        }
        // ---- online softmax (row reductions inside 16-lane halves)
        #pragma unroll
        for (int r = 0; r < 8; ++r) {
            float mx = fmaxf(s[0][r], s[1][r]);
            #pragma unroll
            for (int off = 8; off >= 1; off >>= 1)
                mx = fmaxf(mx, __shfl_xor(mx, off, 32));
            const float mnew  = fmaxf(mrow[r], mx);
            const float alpha = __expf(mrow[r] - mnew);
            mrow[r] = mnew;
            const float p0 = __expf(s[0][r] - mnew);
            const float p1 = __expf(s[1][r] - mnew);
            s[0][r] = p0; s[1][r] = p1;
            float rs = p0 + p1;
            #pragma unroll
            for (int off = 8; off >= 1; off >>= 1)
                rs += __shfl_xor(rs, off, 32);
            lrow[r] = lrow[r] * alpha + rs;
            #pragma unroll
            for (int ft = 0; ft < 8; ++ft) o[ft][r] *= alpha;
        }
        // ---- C-layout -> A-layout transpose of P via per-wave LDS strip
        #pragma unroll
        for (int nt = 0; nt < 2; ++nt)
            #pragma unroll
            for (int r = 0; r < 8; ++r)
                s_p[wave][r + hi * 8][nt * 16 + lh] = (half_t)s[nt][r];
        const half_t* prow = &s_p[wave][lh][0];
        F16x16 up;
        up.f4[0] = *(const v4f*)(prow + hi * 8);
        up.f4[1] = *(const v4f*)(prow + 16 + hi * 8);
        // ---- O += P · V   (V^T so key-dim is contiguous per lane)
        #pragma unroll
        for (int ft = 0; ft < 8; ++ft) {
            const half_t* vptr = VhT + ((size_t)b * FEAT + ft * 16 + lh) * N1_ + k0 + hi * 8;
            F16x16 uv;
            uv.f4[0] = *(const v4f*)(vptr);
            uv.f4[1] = *(const v4f*)(vptr + 16);
            o[ft] = __builtin_amdgcn_wmma_f32_16x16x32_f16(false, up.h16, false, uv.h16,
                                                           (short)0, o[ft], false, false);
        }
    }

    // ---- epilogue: X = O/l + dense_feat  (f16, into per-wave LDS strip)
    #pragma unroll
    for (int ft = 0; ft < 8; ++ft) {
        #pragma unroll
        for (int r = 0; r < 8; ++r) {
            const int m = wave * 16 + r + hi * 8;
            const size_t qr = (size_t)b * N2_ + q0 + m;
            const float val = o[ft][r] / lrow[r] + dfeat[qr * FEAT + ft * 16 + lh];
            s_x[wave][r + hi * 8][ft * 16 + lh] = (half_t)val;
        }
    }

    // ---- fused output projection: out = X @ Wo^T + bo   (WMMA)
    v16h xa[4];
    #pragma unroll
    for (int c = 0; c < 4; ++c) {
        const half_t* xrow = &s_x[wave][lh][0];
        const int o0 = 32 * c + hi * 8;
        F16x16 u;
        u.f4[0] = *(const v4f*)(xrow + o0);
        u.f4[1] = *(const v4f*)(xrow + o0 + 16);
        xa[c] = u.h16;
    }
    #pragma unroll
    for (int fo = 0; fo < 8; ++fo) {
        v8f acc = vzero;
        #pragma unroll
        for (int c = 0; c < 4; ++c) {
            // B[d][f_out] = Wo[f_out][d]: two 8-half runs along d (at +0 and +16) from Woh row
            const half_t* wptr = Woh + (size_t)(fo * 16 + lh) * FEAT + 32 * c + hi * 8;
            F16x16 uw;
            uw.f4[0] = *(const v4f*)(wptr);
            uw.f4[1] = *(const v4f*)(wptr + 16);
            acc = __builtin_amdgcn_wmma_f32_16x16x32_f16(false, xa[c], false, uw.h16,
                                                         (short)0, acc, false, false);
        }
        const float bov = bo[fo * 16 + lh];
        #pragma unroll
        for (int r = 0; r < 8; ++r) {
            const int m = wave * 16 + r + hi * 8;
            const size_t qr = (size_t)b * N2_ + q0 + m;
            out[qr * FEAT + fo * 16 + lh] = acc[r] + bov;
        }
    }
}

extern "C" void kernel_launch(void* const* d_in, const int* in_sizes, int n_in,
                              void* d_out, int out_size, void* d_ws, size_t ws_size,
                              hipStream_t stream)
{
    const float* sparse_xyz  = (const float*)d_in[0];
    const float* sparse_feat = (const float*)d_in[1];
    const float* dense_xyz   = (const float*)d_in[2];
    const float* dense_feat  = (const float*)d_in[3];
    const float* Wq = (const float*)d_in[4];
    const float* bq = (const float*)d_in[5];
    const float* Wk = (const float*)d_in[6];
    const float* bk = (const float*)d_in[7];
    const float* Wv = (const float*)d_in[8];
    const float* bv = (const float*)d_in[9];
    const float* Wo = (const float*)d_in[10];
    const float* bo = (const float*)d_in[11];
    float* out = (float*)d_out;
    (void)in_sizes; (void)n_in; (void)out_size; (void)ws_size;

    char* ws = (char*)d_ws;
    size_t off = 0;
    auto wsalloc = [&](size_t bytes) -> void* {
        void* p = ws + off;
        off += bytes;
        off = (off + 255) & ~(size_t)255;
        return p;
    };
    half_t* Qh  = (half_t*)wsalloc((size_t)B_ * N2_ * FEAT * sizeof(half_t)); // 16.8 MB
    half_t* Kh  = (half_t*)wsalloc((size_t)B_ * N1_ * FEAT * sizeof(half_t)); //  2.1 MB
    half_t* VhT = (half_t*)wsalloc((size_t)B_ * FEAT * N1_ * sizeof(half_t)); //  2.1 MB
    half_t* Woh = (half_t*)wsalloc((size_t)FEAT * FEAT * sizeof(half_t));     //   32 KB

    proj131_kernel<<<B_ * N2_, 128, 0, stream>>>(dense_xyz, dense_feat, Wq, bq, Qh, SCALE_);
    proj131_kernel<<<B_ * N1_, 128, 0, stream>>>(sparse_xyz, sparse_feat, Wk, bk, Kh, 1.0f);
    projv_kernel  <<<B_ * N1_, 128, 0, stream>>>(sparse_feat, Wv, bv, VhT);
    cvt16_kernel  <<<(FEAT * FEAT + 255) / 256, 256, 0, stream>>>(Wo, Woh, FEAT * FEAT);
    attn_kernel   <<<B_ * (N2_ / QT), 128, 0, stream>>>(Qh, Kh, VhT, Woh, bo,
                                                        dense_xyz, sparse_xyz, dense_feat, out);
}
